// Atominator_16449724744302
// MI455X (gfx1250) — matrix operand order repaired and checked by reading, not compiled, not run on hardware
//
#include <hip/hip_runtime.h>
#include <hip/hip_bf16.h>
#include <math.h>

// ---------------------------------------------------------------------------
// CDNA5 (gfx1250) wave32 WMMA types
// ---------------------------------------------------------------------------
typedef __attribute__((ext_vector_type(16))) __bf16 bf16x16;
typedef __attribute__((ext_vector_type(8)))  float  f32x8;

union BF16Frag { unsigned short s[16]; bf16x16 v; };

__device__ inline f32x8 f32x8_zero() {
    f32x8 z;
#pragma unroll
    for (int i = 0; i < 8; ++i) z[i] = 0.f;
    return z;
}

// round-to-nearest-even float -> bf16 (raw bits)
__device__ inline unsigned short f2bf(float f) {
    union { float f; unsigned u; } v; v.f = f;
    unsigned r = v.u + 0x7FFFu + ((v.u >> 16) & 1u);
    return (unsigned short)(r >> 16);
}

// pack two floats into dword of 2 bf16
__device__ inline unsigned pack2bf(float a, float b) {
    return (unsigned)f2bf(a) | ((unsigned)f2bf(b) << 16);
}

// Softplus(beta=0.5): 2*logaddexp(0, 0.5x), numerically stable
__device__ inline float softplus_b05(float x) {
    float a = 0.5f * x;
    return 2.f * (fmaxf(a, 0.f) + log1pf(expf(-fabsf(a))));
}

// ---------------------------------------------------------------------------
// CDNA5 async global->LDS copy (ASYNCcnt path). Builtin signature on this
// toolchain: (V4i addrspace(1)*, V4i addrspace(3)*, imm int, imm int).
// ---------------------------------------------------------------------------
#if defined(__HIP_DEVICE_COMPILE__) && __has_builtin(__builtin_amdgcn_global_load_async_to_lds_b128)
#define ASYNC_LDS 1
#else
#define ASYNC_LDS 0
#endif

#if ASYNC_LDS
typedef int v4i __attribute__((ext_vector_type(4)));
typedef __attribute__((address_space(1))) v4i* gptr_v4i;   // ptr to as(1) int4
typedef __attribute__((address_space(3))) v4i* lptr_v4i;   // ptr to as(3) int4
__device__ inline void async_copy16(const void* g, void* l) {
    __builtin_amdgcn_global_load_async_to_lds_b128((gptr_v4i)g, (lptr_v4i)l, 0, 0);
}
__device__ inline void wait_async0() {
#if __has_builtin(__builtin_amdgcn_s_wait_asynccnt)
    __builtin_amdgcn_s_wait_asynccnt(0);
#else
    asm volatile("s_wait_asynccnt 0x0" ::: "memory");
#endif
}
#endif

// ---------------------------------------------------------------------------
// A-operand loader functors (fp32 virtual matrices, converted to bf16 in LDS)
// ---------------------------------------------------------------------------
struct PlainA {
    const float* p; int ld;    // ld must be a multiple of 4 (16B rows)
    static constexpr bool kVec4 = true;
    __device__ float  ld_elem(int r, int k) const { return p[(size_t)r * ld + k]; }
    __device__ float4 ld_vec4(int r, int k) const {
        return *(const float4*)(p + (size_t)r * ld + k);
    }
};

// pair regressor input x[r] = [node[j0] (512) | node[j1] (512) | rbf (5) | gs (64)]
struct PairA {
    const float* node; const int* jidx; const float* jr; const float* gs;
    static constexpr bool kVec4 = false;   // gather + concat boundaries -> scalar
    __device__ float ld_elem(int r, int k) const {
        if (k < 512)  return node[(size_t)jidx[2 * r] * 512 + k];
        if (k < 1024) return node[(size_t)jidx[2 * r + 1] * 512 + (k - 512)];
        if (k < 1029) return jr[(size_t)r * 8 + (k - 1024)];
        return gs[k - 1029];
    }
    __device__ float4 ld_vec4(int, int) const { return make_float4(0.f, 0.f, 0.f, 0.f); }
};

// ---------------------------------------------------------------------------
// Weight pre-transpose + fp32->bf16 convert: Bt[Npad][Kpad] = B[K,N]^T, zero pad
// ---------------------------------------------------------------------------
__global__ void conv_wt(const float* __restrict__ B, unsigned short* __restrict__ Bt,
                        int K, int N, int Kpad, int Npad)
{
    int i = blockIdx.x * 256 + threadIdx.x;
    if (i < Npad * Kpad) {
        int n = i / Kpad, k = i - n * Kpad;
        float v = (n < N && k < K) ? B[(size_t)k * N + n] : 0.f;
        Bt[i] = f2bf(v);
    }
}

// ---------------------------------------------------------------------------
// Generic WMMA GEMM: C[M,N] = epi(A[M,K] @ B[K,N] + bias)
// B pre-converted bf16, transposed [Npad][Kpad] (k-contiguous rows) so the
// LDS tile is filled by raw async 16B copies (no conversion in hot loop).
// A tile staged via vectorized float4 fast path when block-uniformly in
// bounds, scalar guarded path only for edge tiles / virtual gathers.
// Block tile 64x128: 8 waves = 4 row-tiles x 2 col-groups, each wave 16x64.
// EPI: 0=none 1=softplus(0.5) 2=relu 3=leaky-relu(0.01)
// ---------------------------------------------------------------------------
template <class AL, int EPI>
__global__ __launch_bounds__(256) void wmma_gemm(
    AL A, const unsigned short* __restrict__ Bt, const float* __restrict__ bias,
    float* __restrict__ C, int M, int N, int K, int ldbt, int ldc)
{
    __shared__ unsigned short As[64][32];
    __shared__ unsigned short Bs[128][32];   // [n][k], 64B rows (16B aligned chunks)

    const int tid  = threadIdx.x;
    const int lane = tid & 31;
    const int wave = tid >> 5;
    const int wr   = wave & 3;     // row tile 0..3
    const int wc   = wave >> 2;    // col group 0..1
    const int m0   = blockIdx.x * 64;
    const int n0   = blockIdx.y * 128;

    f32x8 acc[4];
#pragma unroll
    for (int t = 0; t < 4; ++t) acc[t] = f32x8_zero();

    const int hf = lane >> 4;

    for (int k0 = 0; k0 < K; k0 += 32) {
        // stage A tile 64x32 (fp32 -> bf16)
        bool fast = false;
        if constexpr (AL::kVec4) fast = (m0 + 64 <= M) && (k0 + 32 <= K);
        if (fast) {
            // vectorized: 512 float4 chunks, no guards, packed dword LDS stores
            for (int i = tid; i < 512; i += 256) {
                int r = i >> 3, k = (i & 7) << 2;
                float4 v = A.ld_vec4(m0 + r, k0 + k);
                *(uint2*)&As[r][k] = make_uint2(pack2bf(v.x, v.y), pack2bf(v.z, v.w));
            }
        } else {
            for (int i = tid; i < 64 * 32; i += 256) {
                int r = i >> 5, k = i & 31;
                int gm = m0 + r, gk = k0 + k;
                float v = (gm < M && gk < K) ? A.ld_elem(gm, gk) : 0.f;
                As[r][k] = f2bf(v);
            }
        }
        // stage B tile: 128 rows x 32 bf16 = 512 x 16B chunks (async DMA)
#if ASYNC_LDS
        for (int i = tid; i < 512; i += 256) {
            int n = i >> 2, kc = (i & 3) << 3;
            async_copy16(Bt + (size_t)(n0 + n) * ldbt + k0 + kc, &Bs[n][kc]);
        }
        wait_async0();
#else
        for (int i = tid; i < 512; i += 256) {
            int n = i >> 2, kc = (i & 3) << 3;
            *(uint4*)&Bs[n][kc] = *(const uint4*)(Bt + (size_t)(n0 + n) * ldbt + k0 + kc);
        }
#endif
        __syncthreads();

        // A fragment: 16x32, ISA 7.12.2 layout
        BF16Frag af;
        const int mrow = wr * 16 + (lane & 15);
#pragma unroll
        for (int e = 0; e < 16; ++e) {
            int v  = e >> 1;
            int kk = ((v < 4) ? 0 : 16) + hf * 8 + ((v & 3) << 1) + (e & 1);
            af.s[e] = As[mrow][kk];
        }
#pragma unroll
        for (int t = 0; t < 4; ++t) {
            const int nl = wc * 64 + t * 16 + (lane & 15);
            BF16Frag bfr;
#pragma unroll
            for (int e = 0; e < 16; ++e) bfr.s[e] = Bs[nl][((e < 8) ? 0 : 8) + hf * 8 + e];
            acc[t] = __builtin_amdgcn_wmma_f32_16x16x32_bf16(
                false, af.v, false, bfr.v, (short)0, acc[t], false, false);
        }
        __syncthreads();
    }

    // epilogue + store (C layout: VGPR r -> M = r + hf*8, N = lane&15)
#pragma unroll
    for (int t = 0; t < 4; ++t) {
        const int gn = n0 + wc * 64 + t * 16 + (lane & 15);
        float bv = (bias != nullptr && gn < N) ? bias[gn] : 0.f;
#pragma unroll
        for (int r = 0; r < 8; ++r) {
            const int gm = m0 + wr * 16 + hf * 8 + r;
            if (gm < M && gn < N) {
                float v = acc[t][r] + bv;
                if (EPI == 1)      v = softplus_b05(v);
                else if (EPI == 2) v = fmaxf(v, 0.f);
                else if (EPI == 3) v = (v >= 0.f) ? v : 0.01f * v;
                C[(size_t)gm * ldc + gn] = v;
            }
        }
    }
}

// ---------------------------------------------------------------------------
// Fused CFConv edge kernel, templated on feature width D (128/256/384):
//   t   = sp(rbf_e @ We1 + be1)          -> LDS (bf16), 32 edges x D
//   h   = t @ We2 + be2                  -> WMMA, K loop over D
//   msg = new_node[src] * h ; cf[dst] += msg (f32 atomics, hits 192MB L2)
// Block: 256 thr = 8 waves (2 row-tiles x 4 col-groups), 32 edges/block.
// We2 pre-converted bf16 transposed [D][D]; staged with async LDS copies.
// ---------------------------------------------------------------------------
template <int D>
__global__ __launch_bounds__(256) void edge_conv(
    const float* __restrict__ rbf, const int* __restrict__ src,
    const int* __restrict__ dst, const float* __restrict__ newnode,
    const float* __restrict__ We1, const float* __restrict__ be1,
    const unsigned short* __restrict__ We2t, const float* __restrict__ be2,
    float* __restrict__ cf, int E)
{
    constexpr int NT = D / 64;               // n-tiles per wave (2/4/6)
    __shared__ unsigned short tL[32][D];     // t, bf16
    __shared__ unsigned short bL[D][32];     // We2 K-slab, [n][k]

    const int tid  = threadIdx.x;
    const int lane = tid & 31;
    const int wave = tid >> 5;
    const int wr   = wave & 1;      // edge-row tile 0/1
    const int wc   = wave >> 1;     // col group 0..3
    const int e0   = blockIdx.x * 32;
    const int hf   = lane >> 4;

    // phase 1: edge MLP first layer + softplus -> tL
    for (int i = tid; i < 32 * D; i += 256) {
        int r = i / D, c = i - r * D;
        int e = e0 + r;
        float q = 0.f;
        if (e < E) {
            q = be1[c];
            q += rbf[(size_t)e * 8 + 0] * We1[c];
            q += rbf[(size_t)e * 8 + 1] * We1[D + c];
            q += rbf[(size_t)e * 8 + 2] * We1[2 * D + c];
            q += rbf[(size_t)e * 8 + 3] * We1[3 * D + c];
            q += rbf[(size_t)e * 8 + 4] * We1[4 * D + c];
            q = softplus_b05(q);
        }
        tL[r][c] = f2bf(q);
    }
    __syncthreads();

    f32x8 acc[NT];
#pragma unroll
    for (int j = 0; j < NT; ++j) acc[j] = f32x8_zero();

    // phase 2: h = t @ We2 via WMMA, K loop
    for (int k0 = 0; k0 < D; k0 += 32) {
#if ASYNC_LDS
        for (int i = tid; i < D * 4; i += 256) {
            int n = i >> 2, kc = (i & 3) << 3;
            async_copy16(We2t + (size_t)n * D + k0 + kc, &bL[n][kc]);
        }
        wait_async0();
#else
        for (int i = tid; i < D * 4; i += 256) {
            int n = i >> 2, kc = (i & 3) << 3;
            *(uint4*)&bL[n][kc] = *(const uint4*)(We2t + (size_t)n * D + k0 + kc);
        }
#endif
        __syncthreads();

        BF16Frag af;
        const int m = wr * 16 + (lane & 15);
#pragma unroll
        for (int e = 0; e < 16; ++e) {
            int v  = e >> 1;
            int kk = ((v < 4) ? 0 : 16) + hf * 8 + ((v & 3) << 1) + (e & 1);
            af.s[e] = tL[m][k0 + kk];
        }
#pragma unroll
        for (int j = 0; j < NT; ++j) {
            const int n = (wc + (j << 2)) * 16 + (lane & 15);
            BF16Frag bfr;
#pragma unroll
            for (int e = 0; e < 16; ++e) bfr.s[e] = bL[n][((e < 8) ? 0 : 8) + hf * 8 + e];
            acc[j] = __builtin_amdgcn_wmma_f32_16x16x32_bf16(
                false, af.v, false, bfr.v, (short)0, acc[j], false, false);
        }
        __syncthreads();
    }

    // phase 3: +be2, gather-multiply, scatter-add
    int se[8], de[8];
#pragma unroll
    for (int r = 0; r < 8; ++r) {
        int e = e0 + wr * 16 + hf * 8 + r;
        se[r] = (e < E) ? src[e] : 0;
        de[r] = (e < E) ? dst[e] : 0;
    }
#pragma unroll
    for (int j = 0; j < NT; ++j) {
        const int n = (wc + (j << 2)) * 16 + (lane & 15);
        const float bv = be2[n];
#pragma unroll
        for (int r = 0; r < 8; ++r) {
            int e = e0 + wr * 16 + hf * 8 + r;
            if (e < E) {
                float h   = acc[j][r] + bv;
                float msg = newnode[(size_t)se[r] * D + n] * h;
                atomicAdd(&cf[(size_t)de[r] * D + n], msg);
            }
        }
    }
}

// ---------------------------------------------------------------------------
// Elementwise / reduction helper kernels (BW trivial vs 23.3 TB/s)
// ---------------------------------------------------------------------------
__global__ void embed_kernel(const int* __restrict__ nt, const float* __restrict__ emb,
                             float* __restrict__ node_f, int N)
{
    int i = blockIdx.x * 256 + threadIdx.x;
    if (i < N * 128) {
        int r = i >> 7, c = i & 127;
        node_f[(size_t)r * 512 + c] = emb[nt[r] * 128 + c];
    }
}

// RBFLayer(0,5,1): centers j*1.25, exp(-(d-c)^2/1.25); padded to 8 cols
__global__ void rbf_kernel(const float* __restrict__ d, float* __restrict__ out, int M)
{
    int i = blockIdx.x * 256 + threadIdx.x;
    if (i < M * 8) {
        int r = i >> 3, j = i & 7;
        float dd = d[r] - 1.25f * (float)j;
        out[i] = (j < 5) ? expf(-0.8f * dd * dd) : 0.f;
    }
}

__global__ __launch_bounds__(256) void bn_stats(const float* __restrict__ z,
                                                float* __restrict__ mean,
                                                float* __restrict__ rstd, int N)
{
    int c = blockIdx.x;
    __shared__ float sA[256], sB[256];
    float s = 0.f, ss = 0.f;
    for (int r = threadIdx.x; r < N; r += 256) {
        float v = z[(size_t)r * 128 + c]; s += v; ss += v * v;
    }
    sA[threadIdx.x] = s; sB[threadIdx.x] = ss;
    __syncthreads();
    for (int st = 128; st > 0; st >>= 1) {
        if (threadIdx.x < st) { sA[threadIdx.x] += sA[threadIdx.x + st]; sB[threadIdx.x] += sB[threadIdx.x + st]; }
        __syncthreads();
    }
    if (threadIdx.x == 0) {
        float mu = sA[0] / (float)N;
        float var = sB[0] / (float)N - mu * mu;
        mean[c] = mu; rstd[c] = rsqrtf(var + 1e-5f);
    }
}

__global__ void bn_apply(const float* __restrict__ z, const float* __restrict__ mean,
                         const float* __restrict__ rstd, const float* __restrict__ gamma,
                         const float* __restrict__ beta, float* __restrict__ node_f,
                         int N, int coff)
{
    int i = blockIdx.x * 256 + threadIdx.x;
    if (i < N * 128) {
        int r = i >> 7, c = i & 127;
        node_f[(size_t)r * 512 + coff + c] = (z[i] - mean[c]) * rstd[c] * gamma[c] + beta[c];
    }
}

__global__ __launch_bounds__(256) void colsum(const float* __restrict__ node_f,
                                              float* __restrict__ s512, int N)
{
    int c = blockIdx.x;
    __shared__ float sh[256];
    float s = 0.f;
    for (int r = threadIdx.x; r < N; r += 256) s += node_f[(size_t)r * 512 + c];
    sh[threadIdx.x] = s;
    __syncthreads();
    for (int st = 128; st > 0; st >>= 1) {
        if (threadIdx.x < st) sh[threadIdx.x] += sh[threadIdx.x + st];
        __syncthreads();
    }
    if (threadIdx.x == 0) s512[c] = sh[0];
}

__global__ __launch_bounds__(256) void readout_kernel(
    const float* __restrict__ s512, const float* __restrict__ Wr1,
    const float* __restrict__ br1, const float* __restrict__ Wr2,
    const float* __restrict__ br2, float* __restrict__ gs)
{
    __shared__ float hid[256];
    int t = threadIdx.x;
    float a = br1[t];
    for (int k = 0; k < 512; ++k) a += s512[k] * Wr1[k * 256 + t];
    hid[t] = fmaxf(a, 0.f);
    __syncthreads();
    if (t < 64) {
        float g = br2[t];
        for (int k = 0; k < 256; ++k) g += hid[k] * Wr2[k * 64 + t];
        gs[t] = g;
    }
}

// ---------------------------------------------------------------------------
// host-side launchers
// ---------------------------------------------------------------------------
template <int EPI, class AL>
static inline void run_gemm(AL a, const unsigned short* Bt, int ldbt,
                            const float* bias, float* C,
                            int M, int Ncols, int K, int ldc, hipStream_t st)
{
    dim3 grid((M + 63) / 64, (Ncols + 127) / 128);
    wmma_gemm<AL, EPI><<<grid, dim3(256), 0, st>>>(a, Bt, bias, C, M, Ncols, K, ldbt, ldc);
}

static inline void run_conv_wt(const float* B, unsigned short* Bt, int K, int N,
                               int Kpad, int Npad, hipStream_t st)
{
    conv_wt<<<(Npad * Kpad + 255) / 256, 256, 0, st>>>(B, Bt, K, N, Kpad, Npad);
}

extern "C" void kernel_launch(void* const* d_in, const int* in_sizes, int n_in,
                              void* d_out, int out_size, void* d_ws, size_t ws_size,
                              hipStream_t stream)
{
    (void)n_in; (void)out_size; (void)ws_size;
    const int N = in_sizes[0];   // 50000 nodes
    const int E = in_sizes[1];   // 800000 edges
    const int P = in_sizes[5];   // 50000 pairs

    const int*   node_type = (const int*)d_in[0];
    const int*   srcp      = (const int*)d_in[1];
    const int*   dstp      = (const int*)d_in[2];
    const float* edge_dist = (const float*)d_in[3];
    const int*   j_idx     = (const int*)d_in[4];
    const float* j_dist    = (const float*)d_in[5];
    const float* emb       = (const float*)d_in[6];

    // workspace layout (256B aligned)
    char* w = (char*)d_ws; size_t off = 0;
    auto alloc = [&](size_t bytes) -> void* {
        void* p = w + off; off += (bytes + 255) & ~(size_t)255; return p;
    };
    float* node_f  = (float*)alloc((size_t)N * 512 * 4);  // concat node features
    float* newnode = (float*)alloc((size_t)N * 384 * 4);  // node @ W1
    float* cf      = (float*)alloc((size_t)N * 384 * 4);  // scatter-sum target
    float* rbf_e   = (float*)alloc((size_t)E * 8 * 4);    // edge RBF (pad 5->8)
    float* jrbf    = (float*)alloc((size_t)P * 8 * 4);    // pair RBF
    float* zb      = (float*)alloc((size_t)N * 128 * 4);
    float* s1b     = (float*)alloc((size_t)N * 128 * 4);
    float* meanb   = (float*)alloc(128 * 4);
    float* rstdb   = (float*)alloc(128 * 4);
    float* s512    = (float*)alloc(512 * 4);
    float* gsb     = (float*)alloc(64 * 4);
    // pre-converted bf16 transposed weights
    unsigned short* W1t[3];  unsigned short* We2t[3];
    unsigned short* W2t[3];  unsigned short* W3t[3];
    for (int l = 0; l < 3; ++l) {
        const int D = 128 * (l + 1);
        W1t[l]  = (unsigned short*)alloc((size_t)D * D * 2);    // [D][D]
        We2t[l] = (unsigned short*)alloc((size_t)D * D * 2);    // [D][D]
        W2t[l]  = (unsigned short*)alloc((size_t)128 * D * 2);  // [128][D]
        W3t[l]  = (unsigned short*)alloc((size_t)128 * 128 * 2);
    }
    unsigned short* Wg1t = (unsigned short*)alloc((size_t)640 * 1120 * 2); // [640][1120]
    unsigned short* Wg2t = (unsigned short*)alloc((size_t)128 * 576 * 2);  // [128][576]
    unsigned short* Wg3t = (unsigned short*)alloc((size_t)128 * 128 * 2);  // [128][128]
    float* y1 = newnode;  // [P,548] reuse of newnode+cf region post-conv
    float* y2 = zb;       // [P,128] reuse

    // --- weight conversion (tiny, once per call) ---
    for (int l = 0; l < 3; ++l) {
        const int D = 128 * (l + 1);
        run_conv_wt((const float*)d_in[7 + l * 11 + 0], W1t[l],  D, D,   D,   D,   stream);
        run_conv_wt((const float*)d_in[7 + l * 11 + 3], We2t[l], D, D,   D,   D,   stream);
        run_conv_wt((const float*)d_in[7 + l * 11 + 5], W2t[l],  D, 128, D,   128, stream);
        run_conv_wt((const float*)d_in[7 + l * 11 + 7], W3t[l],  128, 128, 128, 128, stream);
    }
    run_conv_wt((const float*)d_in[44], Wg1t, 1093, 546, 1120, 640, stream);
    run_conv_wt((const float*)d_in[46], Wg2t, 546, 128, 576, 128, stream);
    run_conv_wt((const float*)d_in[48], Wg3t, 128, 4,   128, 128, stream);

    embed_kernel<<<(N * 128 + 255) / 256, 256, 0, stream>>>(node_type, emb, node_f, N);
    rbf_kernel<<<(E * 8 + 255) / 256, 256, 0, stream>>>(edge_dist, rbf_e, E);
    rbf_kernel<<<(P * 8 + 255) / 256, 256, 0, stream>>>(j_dist, jrbf, P);

    for (int l = 0; l < 3; ++l) {
        const int D = 128 * (l + 1);
        const float* We1   = (const float*)d_in[7 + l * 11 + 1];
        const float* be1   = (const float*)d_in[7 + l * 11 + 2];
        const float* be2   = (const float*)d_in[7 + l * 11 + 4];
        const float* b2    = (const float*)d_in[7 + l * 11 + 6];
        const float* b3    = (const float*)d_in[7 + l * 11 + 8];
        const float* gamma = (const float*)d_in[7 + l * 11 + 9];
        const float* beta  = (const float*)d_in[7 + l * 11 + 10];

        // new_node = node @ W1
        run_gemm<0>(PlainA{node_f, 512}, W1t[l], D, nullptr, newnode, N, D, D, D, stream);
        // fused edge MLP + gather-mul + scatter-sum
        (void)hipMemsetAsync(cf, 0, (size_t)N * D * 4, stream);
        const int eblocks = (E + 31) / 32;
        if (D == 128)
            edge_conv<128><<<eblocks, 256, 0, stream>>>(rbf_e, srcp, dstp, newnode,
                                                        We1, be1, We2t[l], be2, cf, E);
        else if (D == 256)
            edge_conv<256><<<eblocks, 256, 0, stream>>>(rbf_e, srcp, dstp, newnode,
                                                        We1, be1, We2t[l], be2, cf, E);
        else
            edge_conv<384><<<eblocks, 256, 0, stream>>>(rbf_e, srcp, dstp, newnode,
                                                        We1, be1, We2t[l], be2, cf, E);
        // z = sp(cf @ W2 + b2) @ W3 + b3
        run_gemm<1>(PlainA{cf, D}, W2t[l], D, b2, s1b, N, 128, D, 128, stream);
        run_gemm<0>(PlainA{s1b, 128}, W3t[l], 128, b3, zb, N, 128, 128, 128, stream);
        // BatchNorm (train-mode biased var) + dense-shortcut concat
        bn_stats<<<128, 256, 0, stream>>>(zb, meanb, rstdb, N);
        bn_apply<<<(N * 128 + 255) / 256, 256, 0, stream>>>(zb, meanb, rstdb, gamma,
                                                            beta, node_f, N, 128 * (l + 1));
    }

    // readout: graph_state = relu(sum(node) @ Wr1 + br1) @ Wr2 + br2
    const float* Wr1 = (const float*)d_in[40];
    const float* br1 = (const float*)d_in[41];
    const float* Wr2 = (const float*)d_in[42];
    const float* br2 = (const float*)d_in[43];
    colsum<<<512, 256, 0, stream>>>(node_f, s512, N);
    readout_kernel<<<1, 256, 0, stream>>>(s512, Wr1, br1, Wr2, br2, gsb);

    // pair regressor (x gathered virtually inside the GEMM A-loader)
    const float* bg1 = (const float*)d_in[45];
    const float* bg2 = (const float*)d_in[47];
    const float* bg3 = (const float*)d_in[49];

    // y1 ld padded to 548 so PlainA float4 fast path stays 16B-aligned
    run_gemm<3>(PairA{node_f, j_idx, jrbf, gsb}, Wg1t, 1120, bg1, y1, P, 546, 1093, 548, stream);
    run_gemm<3>(PlainA{y1, 548}, Wg2t, 576, bg2, y2, P, 128, 546, 128, stream);
    run_gemm<0>(PlainA{y2, 128}, Wg3t, 128, bg3, (float*)d_out, P, 4, 128, 4, stream);
}
